// Attention_37056977830181
// MI455X (gfx1250) — matrix operand order repaired and checked by reading, not compiled, not run on hardware
//
#include <hip/hip_runtime.h>
#include <hip/hip_bf16.h>
#include <stdint.h>

// ---------------------------------------------------------------------------
// Attention block for MI455X (gfx1250): bf16 WMMA GEMMs, f32 softmax/accum.
// B=4, N=2048, C=1024, H=16, D=64. wave32, v_wmma_f32_16x16x32_bf16.
// ---------------------------------------------------------------------------

typedef __bf16 bf16;
typedef __bf16   v16bf __attribute__((ext_vector_type(16)));
typedef float    v8f   __attribute__((ext_vector_type(8)));
typedef uint32_t u32x4 __attribute__((ext_vector_type(4)));

union Frag16 { v16bf v; u32x4 q[2]; };

__device__ inline v8f zero8() {
    v8f z;
#pragma unroll
    for (int i = 0; i < 8; ++i) z[i] = 0.0f;
    return z;
}

__device__ inline v8f wmma_bf16(v16bf a, v16bf b, v8f c) {
    // (neg_a, A, neg_b, B, c_mod, C, reuse_a, reuse_b)
    return __builtin_amdgcn_wmma_f32_16x16x32_bf16(false, a, false, b,
                                                   (short)0, c, false, false);
}

// ------------------------------- constants ---------------------------------
constexpr int Bc = 4, Nc = 2048, Cc = 1024, Hc = 16, Dc = 64;
constexpr int Mrows = Bc * Nc;          // 8192
constexpr float SCALE = 0.125f;         // D^-0.5
constexpr float EPS = 1e-6f;

// ---------------------------- f32 -> bf16 ----------------------------------
__global__ void cvt_f32_bf16(const float* __restrict__ src,
                             bf16* __restrict__ dst, int n) {
    int i = blockIdx.x * blockDim.x + threadIdx.x;
    int stride = gridDim.x * blockDim.x;
    for (; i < n; i += stride) dst[i] = (bf16)src[i];
}

// ---------------- QKV GEMM (x @ qkv_w^T) + fused per-head RMSNorm ----------
// Grid: (Mrows/64, 3072/128). WG tile 64x128, 8 waves of 32x32.
// Outputs: q,k -> bf16 [B,H,N,64] (SCALE folded into q); v -> bf16 [B,H,64,N].
__global__ __launch_bounds__(256) void qkv_gemm_norm_kernel(
    const bf16* __restrict__ xb, const bf16* __restrict__ wb,
    const float* __restrict__ qnw, const float* __restrict__ knw,
    bf16* __restrict__ qo, bf16* __restrict__ ko, bf16* __restrict__ vt)
{
    __shared__ float tile[64 * 132];   // 64x128 tile, padded stride

    const int tid  = threadIdx.x;
    const int wid  = tid >> 5;
    const int lane = tid & 31;
    const int lh   = lane >> 4;        // lane half
    const int lm   = lane & 15;

    const int mb = blockIdx.x * 64;    // row base in [0,8192)
    const int nb = blockIdx.y * 128;   // col base in [0,3072)
    const int wm = (wid >> 2) * 32;    // wave row offset (0|32)
    const int wn = (wid & 3) * 32;     // wave col offset (0..96)

    v8f acc00 = zero8(), acc01 = zero8(), acc10 = zero8(), acc11 = zero8();

    const bf16* aptr0 = xb + (size_t)(mb + wm + lm) * Cc;
    const bf16* aptr1 = aptr0 + 16 * Cc;
    const bf16* bptr0 = wb + (size_t)(nb + wn + lm) * Cc;
    const bf16* bptr1 = bptr0 + 16 * Cc;

    for (int kk = 0; kk < Cc; kk += 32) {
        Frag16 a0, a1, b0, b1;
        a0.q[0] = *(const u32x4*)(aptr0 + kk + lh * 8);
        a0.q[1] = *(const u32x4*)(aptr0 + kk + 16 + lh * 8);
        a1.q[0] = *(const u32x4*)(aptr1 + kk + lh * 8);
        a1.q[1] = *(const u32x4*)(aptr1 + kk + 16 + lh * 8);
        b0.q[0] = *(const u32x4*)(bptr0 + kk + lh * 8);
        b0.q[1] = *(const u32x4*)(bptr0 + kk + 16 + lh * 8);
        b1.q[0] = *(const u32x4*)(bptr1 + kk + lh * 8);
        b1.q[1] = *(const u32x4*)(bptr1 + kk + 16 + lh * 8);
        acc00 = wmma_bf16(a0.v, b0.v, acc00);
        acc01 = wmma_bf16(a0.v, b1.v, acc01);
        acc10 = wmma_bf16(a1.v, b0.v, acc10);
        acc11 = wmma_bf16(a1.v, b1.v, acc11);
    }

    // Stage f32 results in LDS for the norm/transpose epilogue.
#pragma unroll
    for (int r = 0; r < 8; ++r) {
        const int rr = (lh ? 8 : 0) + r;
        tile[(wm + rr) * 132 + wn + lm]           = acc00[r];
        tile[(wm + rr) * 132 + wn + 16 + lm]      = acc01[r];
        tile[(wm + 16 + rr) * 132 + wn + lm]      = acc10[r];
        tile[(wm + 16 + rr) * 132 + wn + 16 + lm] = acc11[r];
    }
    __syncthreads();

    const int which = nb / Cc;                 // 0=q, 1=k, 2=v (128 | 1024)
    if (which < 2) {
        const float* nw = (which == 0) ? qnw : knw;
        const float sc  = (which == 0) ? SCALE : 1.0f;
        bf16* dst       = (which == 0) ? qo : ko;
        // 64 rows x 2 head-blocks of 64 = 128 norm rows
        for (int t = tid; t < 128; t += 256) {
            const int r  = t >> 1;
            const int hb = t & 1;
            const float* src = &tile[r * 132 + hb * 64];
            float ss = 0.0f;
#pragma unroll
            for (int d = 0; d < Dc; ++d) { float v = src[d]; ss += v * v; }
            const float inv = rsqrtf(ss * (1.0f / Dc) + EPS) * sc;
            const int m = mb + r;
            const int b = m >> 11, n = m & (Nc - 1);
            const int col0 = (nb & (Cc - 1)) + hb * 64;
            const int h = col0 >> 6;
            bf16* o = dst + ((size_t)((b * Hc + h) * Nc + n)) * Dc;
#pragma unroll
            for (int d = 0; d < Dc; ++d)
                o[d] = (bf16)(src[d] * inv * nw[d]);
        }
    } else {
        // v: transpose-store into [B,H,64,N] so PV B-fragments are contiguous
        for (int e = tid; e < 64 * 128; e += 256) {
            const int r = e >> 7, c = e & 127;
            const int m = mb + r;
            const int b = m >> 11, n = m & (Nc - 1);
            const int col = (nb & (Cc - 1)) + c;
            const int h = col >> 6, d = col & 63;
            vt[((size_t)((b * Hc + h) * Dc + d)) * Nc + n] = (bf16)tile[r * 132 + c];
        }
    }
}

// ------------------------------- attention ---------------------------------
// Grid: (B*H, N/16). One WG handles 16 queries x 2048 keys.
// LDS: scores f32 [16][2048] (128KB) + P bf16 [16][2048] (64KB) + O partial.
__global__ __launch_bounds__(256) void attn_kernel(
    const bf16* __restrict__ qptr, const bf16* __restrict__ kptr,
    const bf16* __restrict__ vptr, bf16* __restrict__ ao)
{
    extern __shared__ char smem[];
    float* sS = (float*)smem;                              // [16][2048] f32
    bf16*  sP = (bf16*)(smem + 16 * 2048 * 4);             // [16][2048] bf16
    float* sO = (float*)(smem + 16 * 2048 * 4 + 16 * 2048 * 2); // [2][16*64]

    const int bh  = blockIdx.x;      // b*16 + h
    const int qt  = blockIdx.y;      // query tile
    const int tid = threadIdx.x;
    const int wid = tid >> 5, lane = tid & 31;
    const int lh  = lane >> 4, lm = lane & 15;

    const bf16* qbase = qptr + ((size_t)bh * Nc + qt * 16) * Dc;
    const bf16* kbase = kptr + (size_t)bh * Nc * Dc;
    const bf16* vbase = vptr + (size_t)bh * Dc * Nc;

    // Q A-fragments (rows = 16 queries, K = d in [0,64)), loaded once.
    Frag16 qa0, qa1;
    {
        const bf16* qrow = qbase + lm * Dc;
        qa0.q[0] = *(const u32x4*)(qrow + lh * 8);
        qa0.q[1] = *(const u32x4*)(qrow + 16 + lh * 8);
        qa1.q[0] = *(const u32x4*)(qrow + 32 + lh * 8);
        qa1.q[1] = *(const u32x4*)(qrow + 48 + lh * 8);
    }

    // ---- scores: S = (q*scale) @ k^T, 128 key tiles across 8 waves ----
    for (int kt = wid; kt < Nc / 16; kt += 8) {
        const bf16* krow = kbase + (size_t)(kt * 16 + lm) * Dc;
        Frag16 kf0, kf1;
        kf0.q[0] = *(const u32x4*)(krow + lh * 8);
        kf0.q[1] = *(const u32x4*)(krow + 16 + lh * 8);
        kf1.q[0] = *(const u32x4*)(krow + 32 + lh * 8);
        kf1.q[1] = *(const u32x4*)(krow + 48 + lh * 8);
        v8f s = zero8();
        s = wmma_bf16(qa0.v, kf0.v, s);
        s = wmma_bf16(qa1.v, kf1.v, s);
#pragma unroll
        for (int r = 0; r < 8; ++r)
            sS[((lh ? 8 : 0) + r) * Nc + kt * 16 + lm] = s[r];
    }
    __syncthreads();

    // ---- softmax in f32; each wave owns two rows ----
    for (int r2 = 0; r2 < 2; ++r2) {
        const int row = wid * 2 + r2;
        float* srow = sS + row * Nc;
        float mx = -3.4e38f;
        for (int i = lane; i < Nc; i += 32) mx = fmaxf(mx, srow[i]);
#pragma unroll
        for (int off = 16; off; off >>= 1) mx = fmaxf(mx, __shfl_xor(mx, off, 32));
        float sum = 0.0f;
        for (int i = lane; i < Nc; i += 32) {
            const float e = __expf(srow[i] - mx);
            srow[i] = e;
            sum += e;
        }
#pragma unroll
        for (int off = 16; off; off >>= 1) sum += __shfl_xor(sum, off, 32);
        const float inv = 1.0f / sum;
        bf16* prow = sP + row * Nc;
        for (int i = lane; i < Nc; i += 32) prow[i] = (bf16)(srow[i] * inv);
    }
    __syncthreads();

    // ---- O = P @ V : 4 d-tiles x split-K(2) across 8 waves ----
    const int dt = wid & 3;      // 16-wide d tile
    const int kh = wid >> 2;     // key half: [0,1024) or [1024,2048)
    v8f o = zero8();
    const bf16* vrow = vbase + (size_t)(dt * 16 + lm) * Nc + kh * 1024;
    const bf16* pp0  = sP + lm * Nc + kh * 1024;
    for (int kb = 0; kb < 1024; kb += 32) {
        Frag16 pa, vf;
        pa.q[0] = *(const u32x4*)(pp0 + kb + lh * 8);
        pa.q[1] = *(const u32x4*)(pp0 + kb + 16 + lh * 8);
        vf.q[0] = *(const u32x4*)(vrow + kb + lh * 8);
        vf.q[1] = *(const u32x4*)(vrow + kb + 16 + lh * 8);
        o = wmma_bf16(pa.v, vf.v, o);
    }
#pragma unroll
    for (int r = 0; r < 8; ++r)
        sO[kh * 1024 + ((lh ? 8 : 0) + r) * 64 + dt * 16 + lm] = o[r];
    __syncthreads();

    // combine split-K halves, store to attn-out [B,N,C] as bf16
    const int b = bh >> 4, h = bh & 15;
    for (int e = tid; e < 16 * 64; e += 256) {
        const int m = e >> 6, d = e & 63;
        const float v = sO[e] + sO[1024 + e];
        const int n = qt * 16 + m;
        ao[((size_t)(b * Nc + n)) * Cc + h * 64 + d] = (bf16)v;
    }
}

// ------------------- projection GEMM: out = A @ W^T + bias -----------------
__global__ __launch_bounds__(256) void proj_gemm_kernel(
    const bf16* __restrict__ ab, const bf16* __restrict__ wb,
    const float* __restrict__ bias, float* __restrict__ out)
{
    const int tid  = threadIdx.x;
    const int wid  = tid >> 5, lane = tid & 31;
    const int lh   = lane >> 4, lm = lane & 15;
    const int mb   = blockIdx.x * 64;
    const int nb   = blockIdx.y * 128;
    const int wm   = (wid >> 2) * 32;
    const int wn   = (wid & 3) * 32;

    v8f acc00 = zero8(), acc01 = zero8(), acc10 = zero8(), acc11 = zero8();

    const bf16* aptr0 = ab + (size_t)(mb + wm + lm) * Cc;
    const bf16* aptr1 = aptr0 + 16 * Cc;
    const bf16* bptr0 = wb + (size_t)(nb + wn + lm) * Cc;
    const bf16* bptr1 = bptr0 + 16 * Cc;

    for (int kk = 0; kk < Cc; kk += 32) {
        Frag16 a0, a1, b0, b1;
        a0.q[0] = *(const u32x4*)(aptr0 + kk + lh * 8);
        a0.q[1] = *(const u32x4*)(aptr0 + kk + 16 + lh * 8);
        a1.q[0] = *(const u32x4*)(aptr1 + kk + lh * 8);
        a1.q[1] = *(const u32x4*)(aptr1 + kk + 16 + lh * 8);
        b0.q[0] = *(const u32x4*)(bptr0 + kk + lh * 8);
        b0.q[1] = *(const u32x4*)(bptr0 + kk + 16 + lh * 8);
        b1.q[0] = *(const u32x4*)(bptr1 + kk + lh * 8);
        b1.q[1] = *(const u32x4*)(bptr1 + kk + 16 + lh * 8);
        acc00 = wmma_bf16(a0.v, b0.v, acc00);
        acc01 = wmma_bf16(a0.v, b1.v, acc01);
        acc10 = wmma_bf16(a1.v, b0.v, acc10);
        acc11 = wmma_bf16(a1.v, b1.v, acc11);
    }

#pragma unroll
    for (int r = 0; r < 8; ++r) {
        const int rr = (lh ? 8 : 0) + r;
        const int row0 = mb + wm + rr;
        const int row1 = mb + wm + 16 + rr;
        const int c0 = nb + wn + lm;
        const int c1 = nb + wn + 16 + lm;
        out[(size_t)row0 * Cc + c0] = acc00[r] + bias[c0];
        out[(size_t)row0 * Cc + c1] = acc01[r] + bias[c1];
        out[(size_t)row1 * Cc + c0] = acc10[r] + bias[c0];
        out[(size_t)row1 * Cc + c1] = acc11[r] + bias[c1];
    }
}

// ------------------------------- launcher ----------------------------------
extern "C" void kernel_launch(void* const* d_in, const int* in_sizes, int n_in,
                              void* d_out, int out_size, void* d_ws, size_t ws_size,
                              hipStream_t stream) {
    const float* x      = (const float*)d_in[0];
    const float* qkv_w  = (const float*)d_in[1];
    const float* qnw    = (const float*)d_in[2];
    const float* knw    = (const float*)d_in[3];
    const float* proj_w = (const float*)d_in[4];
    const float* proj_b = (const float*)d_in[5];
    float* out = (float*)d_out;

    char* ws = (char*)d_ws;
    bf16* xb    = (bf16*)(ws);                       // 8192*1024   bf16 (16 MB)
    bf16* wqkv  = (bf16*)(ws + 16777216);            // 3072*1024   bf16 ( 6 MB)
    bf16* wproj = (bf16*)(ws + 23068672);            // 1024*1024   bf16 ( 2 MB)
    bf16* qbuf  = (bf16*)(ws + 25165824);            // [B,H,N,64]  bf16 (16 MB)
    bf16* kbuf  = (bf16*)(ws + 41943040);            // [B,H,N,64]  bf16 (16 MB)
    bf16* vtb   = (bf16*)(ws + 58720256);            // [B,H,64,N]  bf16 (16 MB)
    bf16* aob   = (bf16*)(ws + 75497472);            // [B,N,C]     bf16 (16 MB)

    cvt_f32_bf16<<<2048, 256, 0, stream>>>(x,      xb,    Mrows * Cc);
    cvt_f32_bf16<<<2048, 256, 0, stream>>>(qkv_w,  wqkv,  3 * Cc * Cc);
    cvt_f32_bf16<<<1024, 256, 0, stream>>>(proj_w, wproj, Cc * Cc);

    qkv_gemm_norm_kernel<<<dim3(Mrows / 64, (3 * Cc) / 128), 256, 0, stream>>>(
        xb, wqkv, qnw, knw, qbuf, kbuf, vtb);

    const size_t attn_lds = 16 * 2048 * 4 + 16 * 2048 * 2 + 2 * 16 * 64 * 4; // 204800 B
    hipFuncSetAttribute((const void*)attn_kernel,
                        hipFuncAttributeMaxDynamicSharedMemorySize, (int)attn_lds);
    attn_kernel<<<dim3(Bc * Hc, Nc / 16), 256, attn_lds, stream>>>(
        qbuf, kbuf, vtb, aob);

    proj_gemm_kernel<<<dim3(Mrows / 64, Cc / 128), 256, 0, stream>>>(
        aob, wproj, proj_b, out);
}
